// LawhMemoryStore_19370302505126
// MI455X (gfx1250) — matrix operand order repaired and checked by reading, not compiled, not run on hardware
//
#include <hip/hip_runtime.h>
#include <hip/hip_bf16.h>

typedef __bf16 bf16_t;
typedef __attribute__((ext_vector_type(16))) __bf16 v16bf;
typedef __attribute__((ext_vector_type(8)))  __bf16 v8bf;
typedef __attribute__((ext_vector_type(8)))  float  v8f;

#define DIMS 256
#define NQ 2048
#define NP 100000
#define TOPK 8
#define NSLICE 32
#define KEY_TILES 6250        // 100000 / 16 exactly
#define TILES_PER_SLICE 196   // ceil(6250/32)
#define NEG_INF (-__FLT_MAX__)

// ---------------------------------------------------------------------------
// L2-normalize each 256-wide row of `in` and emit bf16. One block per row.
// ---------------------------------------------------------------------------
__global__ __launch_bounds__(256)
void nrm_rows_bf16(const float* __restrict__ in, bf16_t* __restrict__ out) {
  __shared__ float red[256];
  const int r = blockIdx.x, t = threadIdx.x;
  float x = in[(size_t)r * DIMS + t];
  red[t] = x * x;
  __syncthreads();
  for (int s = 128; s > 0; s >>= 1) {
    if (t < s) red[t] += red[t + s];
    __syncthreads();
  }
  float scale = 1.0f / fmaxf(sqrtf(red[0]), 1e-12f);
  out[(size_t)r * DIMS + t] = (bf16_t)(x * scale);
}

// ---------------------------------------------------------------------------
// Fused bf16 WMMA GEMM + streaming top-8.
// grid = (128 query tiles, 32 key slices), block = 1 wave (32 lanes).
// A (16x32 bf16) per-lane layout:  lanes<16 K=32f..32f+15,
//                                  lanes>=16 K=32f+16..32f+31
// B (32x16 bf16) per-lane layout:  col N = lane&15,
//                                  lanes<16 K=32f..32f+15, lanes>=16 +16
// C (16x16 f32):  VGPR r holds row M = r + 8*(lane>>4), col N = lane&15
// ---------------------------------------------------------------------------
__global__ __launch_bounds__(32)
void topk_partial(const bf16_t* __restrict__ qn, const bf16_t* __restrict__ kn,
                  float* __restrict__ pvals, int* __restrict__ pidx) {
  __shared__ float lds_sim[16 * 16];
  const int lane  = threadIdx.x;
  const int qtile = blockIdx.x;
  const int slice = blockIdx.y;
  const int half  = lane >> 4;
  const int l16   = lane & 15;

  // Preload the 8 A fragments (this wave's 16 query rows, full K=256).
  const bf16_t* qrow = qn + (size_t)(qtile * 16 + l16) * DIMS;
  v16bf afrag[8];
#pragma unroll
  for (int f = 0; f < 8; ++f) {
    const bf16_t* p0 = qrow + 32 * f + 16 * half;
    v8bf lo = *(const v8bf*)(p0);
    v8bf hi = *(const v8bf*)(p0 + 8);
    v16bf a;
#pragma unroll
    for (int i = 0; i < 8; ++i) { a[i] = lo[i]; a[i + 8] = hi[i]; }
    afrag[f] = a;
  }

  // Per-lane (lanes 0..15) sorted-descending top-8 for query row l16.
  float tv[TOPK]; int tix[TOPK];
#pragma unroll
  for (int j = 0; j < TOPK; ++j) { tv[j] = NEG_INF; tix[j] = 0x7fffffff; }

  const int t0 = slice * TILES_PER_SLICE;
  const int t1 = min(t0 + TILES_PER_SLICE, KEY_TILES);

  for (int t = t0; t < t1; ++t) {
    const int key = t * 16 + l16;
    const bf16_t* krow = kn + (size_t)key * DIMS + 16 * half;

    v8f c = {};
#pragma unroll
    for (int f = 0; f < 8; ++f) {
      v16bf b = *(const v16bf*)(krow + 32 * f);
      c = __builtin_amdgcn_wmma_f32_16x16x32_bf16(
              /*neg_a=*/false, afrag[f], /*neg_b=*/false, b,
              /*c_mod=*/(short)0, c, /*reuse_a=*/false, /*reuse_b=*/false);
    }
    if (t + 1 < t1) __builtin_prefetch(krow + 16 * DIMS, 0, 0);

    // Stage the 16x16 sim tile: row M = r + 8*half, col N = l16.
#pragma unroll
    for (int r = 0; r < 8; ++r)
      lds_sim[(r + 8 * half) * 16 + l16] = c[r];
    __syncthreads();

    if (half == 0) {  // lane q = l16 owns query row l16
      const int kbase = t * 16;
#pragma unroll 4
      for (int n = 0; n < 16; ++n) {
        float v = lds_sim[l16 * 16 + n];
        if (v > tv[TOPK - 1]) {
          int ix = kbase + n;
#pragma unroll
          for (int j = 0; j < TOPK; ++j) {
            if (v > tv[j]) {
              float fv = tv[j]; int fi = tix[j];
              tv[j] = v; tix[j] = ix; v = fv; ix = fi;
            }
          }
        }
      }
    }
    __syncthreads();  // protect lds_sim against next tile's writes
  }

  if (half == 0) {
    const int q = qtile * 16 + l16;
    const size_t base = ((size_t)q * NSLICE + slice) * TOPK;
#pragma unroll
    for (int j = 0; j < TOPK; ++j) { pvals[base + j] = tv[j]; pidx[base + j] = tix[j]; }
  }
}

// ---------------------------------------------------------------------------
// Merge 32 partial top-8 lists per query into the final top-8 indices.
// Tie-break: smaller key index wins (matches jax.lax.top_k).
// ---------------------------------------------------------------------------
__global__ __launch_bounds__(32)
void topk_merge(const float* __restrict__ pvals, const int* __restrict__ pidx,
                int* __restrict__ out_idx) {
  __shared__ float sv[NSLICE * TOPK];
  __shared__ int   si[NSLICE * TOPK];
  const int q = blockIdx.x, lane = threadIdx.x;
  const size_t base = (size_t)q * NSLICE * TOPK;
  for (int i = lane; i < NSLICE * TOPK; i += 32) {
    sv[i] = pvals[base + i];
    si[i] = pidx[base + i];
  }
  __syncthreads();
  if (lane == 0) {
    float tv[TOPK]; int tix[TOPK];
#pragma unroll
    for (int j = 0; j < TOPK; ++j) { tv[j] = NEG_INF; tix[j] = 0x7fffffff; }
    for (int c = 0; c < NSLICE * TOPK; ++c) {
      float v = sv[c]; int ix = si[c];
      if (v > tv[TOPK - 1] || (v == tv[TOPK - 1] && ix < tix[TOPK - 1])) {
#pragma unroll
        for (int j = 0; j < TOPK; ++j) {
          if (v > tv[j] || (v == tv[j] && ix < tix[j])) {
            float fv = tv[j]; int fi = tix[j];
            tv[j] = v; tix[j] = ix; v = fv; ix = fi;
          }
        }
      }
    }
#pragma unroll
    for (int j = 0; j < TOPK; ++j) out_idx[(size_t)q * TOPK + j] = tix[j];
  }
}

// ---------------------------------------------------------------------------
// Gather K_sel / V_sel rows (f32, exact) for the selected indices.
// One 256-thread block per selected row.
// ---------------------------------------------------------------------------
__global__ __launch_bounds__(256)
void gather_rows(const int* __restrict__ out_idx, const float* __restrict__ keys,
                 const float* __restrict__ values, float* __restrict__ Ksel,
                 float* __restrict__ Vsel) {
  const int g = blockIdx.x, t = threadIdx.x;
  const int idx = out_idx[g];
  Ksel[(size_t)g * DIMS + t] = keys[(size_t)idx * DIMS + t];
  Vsel[(size_t)g * DIMS + t] = values[(size_t)idx * DIMS + t];
}

extern "C" void kernel_launch(void* const* d_in, const int* in_sizes, int n_in,
                              void* d_out, int out_size, void* d_ws, size_t ws_size,
                              hipStream_t stream) {
  const float* queries = (const float*)d_in[0];
  const float* keys    = (const float*)d_in[1];
  const float* values  = (const float*)d_in[2];
  // d_in[3] is k == 8 (compile-time TOPK)

  char* ws = (char*)d_ws;
  size_t off = 0;
  bf16_t* qn = (bf16_t*)(ws + off); off += (size_t)NQ * DIMS * sizeof(bf16_t);
  off = (off + 255) & ~(size_t)255;
  bf16_t* kn = (bf16_t*)(ws + off); off += (size_t)NP * DIMS * sizeof(bf16_t);
  off = (off + 255) & ~(size_t)255;
  float* pvals = (float*)(ws + off); off += (size_t)NQ * NSLICE * TOPK * sizeof(float);
  off = (off + 255) & ~(size_t)255;
  int* pidx = (int*)(ws + off);

  // d_out layout: [ topk_idx int32 (2048*8) | K_sel f32 (16384*256) | V_sel f32 ]
  int*   out_idx = (int*)d_out;
  float* Ksel = (float*)d_out + (size_t)NQ * TOPK;
  float* Vsel = Ksel + (size_t)NQ * TOPK * DIMS;

  nrm_rows_bf16<<<NQ, 256, 0, stream>>>(queries, qn);
  nrm_rows_bf16<<<NP, 256, 0, stream>>>(keys, kn);

  dim3 grid(NQ / 16, NSLICE);
  topk_partial<<<grid, 32, 0, stream>>>(qn, kn, pvals, pidx);

  topk_merge<<<NQ, 32, 0, stream>>>(pvals, pidx, out_idx);
  gather_rows<<<NQ * TOPK, 256, 0, stream>>>(out_idx, keys, values, Ksel, Vsel);
}
